// ProtoNetwork_28458453303759
// MI455X (gfx1250) — compile-verified, matrix operand order checked by reading, not run on hardware
//
#include <hip/hip_runtime.h>
#include <hip/hip_bf16.h>
#include <cstdint>
#include <cstddef>

typedef __attribute__((ext_vector_type(16))) _Float16 v16h;
typedef __attribute__((ext_vector_type(8)))  float    v8f;
typedef __attribute__((ext_vector_type(2)))  float    v2f;
typedef __attribute__((ext_vector_type(4)))  unsigned int v4u;
typedef __attribute__((ext_vector_type(8)))  int      v8i;
typedef __attribute__((ext_vector_type(4)))  int      v4i;

#define NIMG  400
#define NSUP  100
#define NCH   64
#define DFEAT 1600
#define BN_EPS 1e-5f

#if defined(__has_builtin)
#  if __has_builtin(__builtin_amdgcn_tensor_load_to_lds)
#    define HAVE_TDM 1
#  endif
#  if __has_builtin(__builtin_amdgcn_s_wait_tensorcnt)
#    define HAVE_TCNT 1
#  endif
#endif

__device__ inline void wait_tensorcnt0() {
#if defined(HAVE_TCNT)
  __builtin_amdgcn_s_wait_tensorcnt(0);
#elif defined(HAVE_TDM)
  asm volatile("s_wait_tensorcnt 0" ::: "memory");
#endif
}

// ---------------------------------------------------------------------------
// 1. Convert fp32 inputs (support then sample) into one PADDED f16 buffer
//    [img][c][86][86] with a zero 1-pixel halo.
// ---------------------------------------------------------------------------
__global__ void cvt_inputs(const float* __restrict__ sup,
                           const float* __restrict__ smp,
                           _Float16* __restrict__ out) {
  const size_t total = (size_t)NIMG * 3 * 86 * 86;
  size_t idx = (size_t)blockIdx.x * blockDim.x + threadIdx.x;
  if (idx >= total) return;
  int ix = (int)(idx % 86); size_t t = idx / 86;
  int iy = (int)(t % 86);   t /= 86;
  int c  = (int)(t % 3);    int img = (int)(t / 3);
  float v = 0.f;
  if (ix >= 1 && ix <= 84 && iy >= 1 && iy <= 84) {
    int simg = (img < NSUP) ? img : img - NSUP;
    size_t src = ((size_t)simg * 3 + c) * (84 * 84) + (size_t)(iy - 1) * 84 + (ix - 1);
    v = (img < NSUP) ? sup[src] : smp[src];
  }
  out[idx] = (_Float16)v;
}

// ---------------------------------------------------------------------------
// 2. Pack conv weights into per-lane WMMA A-fragment order (f16).
//    index = ((cb*KC + kc)*32 + lane)*16 + h
//    M = lane & 15,  K = kc*32 + (h<8 ? h : h+8) + (lane<16 ? 0 : 8)
// ---------------------------------------------------------------------------
__global__ void pack_w(const float* __restrict__ w, _Float16* __restrict__ wpk,
                       int cin, int KC) {
  int total = 4 * KC * 32 * 16;
  int idx = blockIdx.x * blockDim.x + threadIdx.x;
  if (idx >= total) return;
  int h = idx & 15;
  int t = idx >> 4;
  int lane = t & 31; t >>= 5;
  int kc = t % KC;
  int cb = t / KC;
  int m  = lane & 15;
  int k  = kc * 32 + (h < 8 ? h : h + 8) + (lane < 16 ? 0 : 8);
  float v = 0.f;
  if (k < cin * 9) v = w[(size_t)(cb * 16 + m) * (cin * 9) + k];
  wpk[idx] = (_Float16)v;
}

__global__ void zero_f32(float* __restrict__ p, int n) {
  int i = blockIdx.x * blockDim.x + threadIdx.x;
  if (i < n) p[i] = 0.f;
}

// ---------------------------------------------------------------------------
// 3. Implicit-GEMM conv (3x3 over pre-padded input) + bias + ReLU.
//    Block = 128 threads = 4 waves. Wave w owns pixel subtile w (16 pixels)
//    and loops over all 4 cout blocks (4 wmma / chunk, B reused from regs).
//    Input patch (cin x nrows x (HC+2)) is loaded once to LDS via the TDM.
// ---------------------------------------------------------------------------
template <int HC, int CIN, int KC>
__global__ __launch_bounds__(128) void conv_wmma(
    const _Float16* __restrict__ inP, const _Float16* __restrict__ wpk,
    const float* __restrict__ bias, _Float16* __restrict__ act,
    float* __restrict__ stats) {
  constexpr int WP   = HC + 2;                 // padded width
  constexpr int NPIX = HC * HC;
  constexpr int KMAX = CIN * 9;
  constexpr int NRC  = (63 / HC) + 4;          // max rows spanned by 64 pixels
  constexpr int NR   = (NRC < WP) ? NRC : WP;  // patch row capacity
  __shared__ _Float16 patch[CIN * NR * WP];
  __shared__ float sSum[NCH];
  __shared__ float sSsq[NCH];

  const int tid  = threadIdx.x;
  const int lane = tid & 31;
  const int wv   = tid >> 5;                   // wave = pixel subtile 0..3
  const int img  = blockIdx.y;
  const int pixBase = blockIdx.x * 64;

  int rows0    = pixBase / HC;
  int rowsLast = (pixBase + 63) / HC;
  if (rowsLast > HC - 1) rowsLast = HC - 1;
  const int nrows = rowsLast - rows0 + 3;      // input rows incl. 3x3 halo

  if (tid < NCH) { sSum[tid] = 0.f; sSsq[tid] = 0.f; }

  // ---- load input patch to LDS (TDM) ----
  const _Float16* gbase = inP + ((size_t)img * CIN) * (WP * WP) + (size_t)rows0 * WP;
#if defined(HAVE_TDM)
  if (wv == 0) {
    uint64_t ga = (uint64_t)(uintptr_t)gbase;          // byte address
    uint32_t la = (uint32_t)(uintptr_t)(&patch[0]);    // LDS byte offset
    uint32_t td0 = (uint32_t)(WP * WP);                // tensor dim0 (channel)
    uint32_t td1 = (uint32_t)CIN;
    uint32_t t0  = (uint32_t)(nrows * WP);             // tile dim0
    uint32_t t1  = (uint32_t)CIN;                      // tile dim1
    uint64_t s0  = (uint64_t)(WP * WP);                // dim0 stride (elems)
    v4u g0;
    g0[0] = 1u;                                        // count=1, user mode
    g0[1] = la;                                        // lds_addr
    g0[2] = (uint32_t)ga;                              // global_addr[31:0]
    g0[3] = (uint32_t)((ga >> 32) & 0x01FFFFFFu) | (2u << 30); // addr[56:32]|type=2
    v8i g1;
    g1[0] = (int)(1u << 16);                           // wg_mask=0, data_size=2B
    g1[1] = (int)((td0 & 0xFFFFu) << 16);              // tensor_dim0[15:0]
    g1[2] = (int)((td0 >> 16) | ((td1 & 0xFFFFu) << 16)); // td0 hi | td1 lo
    g1[3] = (int)((td1 >> 16) | (t0 << 16));           // td1 hi | tile_dim0
    g1[4] = (int)(t1 & 0xFFFFu);                       // tile_dim1 | tile_dim2=0
    g1[5] = (int)(uint32_t)s0;                         // stride0[31:0]
    g1[6] = (int)(uint32_t)((s0 >> 32) & 0xFFFFu);     // stride0[47:32]|stride1 lo=0
    g1[7] = 0;
    v4i g2 = {0, 0, 0, 0};
    v4i g3 = {0, 0, 0, 0};
    v8i g4 = {0, 0, 0, 0, 0, 0, 0, 0};
    __builtin_amdgcn_tensor_load_to_lds(g0, g1, g2, g3, g4, 0);
    wait_tensorcnt0();
  }
#else
  const int patchElems = CIN * nrows * WP;
  for (int i = tid; i < patchElems; i += 128) {
    int ch = i / (nrows * WP);
    int r  = i - ch * (nrows * WP);
    patch[i] = gbase[(size_t)ch * (WP * WP) + r];
  }
#endif
  __syncthreads();

  // ---- per-lane invariants for B-fragment im2col gather ----
  const int n   = lane & 15;
  const int pix = pixBase + wv * 16 + n;
  const int pixc = (pix < NPIX) ? pix : (NPIX - 1);
  const int py = pixc / HC;
  const int px = pixc - py * HC;
  const int rowRel = py - rows0;               // 0..nrows-3
  const int kHalf = (lane < 16) ? 0 : 8;

  v8f acc0 = {}, acc1 = {}, acc2 = {}, acc3 = {};
  for (int kc = 0; kc < KC; ++kc) {
    // build B fragment in registers from LDS patch
    v16h bfrag;
    #pragma unroll
    for (int h = 0; h < 16; ++h) {
      const int klocal = (h < 8 ? h : h + 8) + kHalf;
      const int kg = kc * 32 + klocal;
      _Float16 v = (_Float16)0.f;
      if (!(KMAX < KC * 32) || (kg < KMAX)) {
        const int ci  = kg / 9;
        const int rem = kg - ci * 9;
        const int ry  = rem / 3;
        const int rx  = rem - ry * 3;
        v = patch[(ci * nrows + rowRel + ry) * WP + px + rx];
      }
      bfrag[h] = v;
    }
    // 4 cout blocks share this B fragment
    const size_t aoff = ((size_t)kc * 32 + lane) * 16;
    const size_t astep = (size_t)KC * 32 * 16;
    v16h a0 = *(const v16h*)(wpk + aoff);
    v16h a1 = *(const v16h*)(wpk + astep + aoff);
    v16h a2 = *(const v16h*)(wpk + 2 * astep + aoff);
    v16h a3 = *(const v16h*)(wpk + 3 * astep + aoff);
    acc0 = __builtin_amdgcn_wmma_f32_16x16x32_f16(false, a0, false, bfrag, (short)0, acc0, false, false);
    acc1 = __builtin_amdgcn_wmma_f32_16x16x32_f16(false, a1, false, bfrag, (short)0, acc1, false, false);
    acc2 = __builtin_amdgcn_wmma_f32_16x16x32_f16(false, a2, false, bfrag, (short)0, acc2, false, false);
    acc3 = __builtin_amdgcn_wmma_f32_16x16x32_f16(false, a3, false, bfrag, (short)0, acc3, false, false);
  }

  // ---- epilogue: bias + relu + f16 store + BN stats ----
  const int mAdd = (lane < 16) ? 0 : 8;
  if (pix < NPIX) {
    v8f* accs[4] = { &acc0, &acc1, &acc2, &acc3 };
    #pragma unroll
    for (int cb = 0; cb < 4; ++cb) {
      #pragma unroll
      for (int r = 0; r < 8; ++r) {
        int cout = cb * 16 + r + mAdd;
        float val = (*accs[cb])[r] + bias[cout];
        val = val > 0.f ? val : 0.f;
        act[(size_t)(img * NCH + cout) * NPIX + pix] = (_Float16)val;
        atomicAdd(&sSum[cout], val);
        atomicAdd(&sSsq[cout], val * val);
      }
    }
  }
  __syncthreads();
  if (tid < NCH) {
    int g = (img < NSUP) ? 0 : 1;
    atomicAdd(&stats[g * 128 + tid], sSum[tid]);
    atomicAdd(&stats[g * 128 + NCH + tid], sSsq[tid]);
  }
}

// ---------------------------------------------------------------------------
// 4. Finalize BN stats -> per (group, channel) scale/shift
// ---------------------------------------------------------------------------
__global__ void bn_finalize(const float* __restrict__ stats,
                            const float* __restrict__ gamma,
                            const float* __restrict__ beta,
                            float* __restrict__ bnp, int npix) {
  int t = threadIdx.x;
  if (t >= 128) return;
  int g = t >> 6, c = t & 63;
  float count = (g ? 300.f : 100.f) * (float)npix;
  float mean  = stats[g * 128 + c] / count;
  float var   = stats[g * 128 + NCH + c] / count - mean * mean;
  float scale = gamma[c] * rsqrtf(var + BN_EPS);
  bnp[(g * 64 + c) * 2 + 0] = scale;
  bnp[(g * 64 + c) * 2 + 1] = beta[c] - mean * scale;
}

// ---------------------------------------------------------------------------
// 5a. BN apply + 2x2 maxpool -> PADDED f16 output (zero halo materialized)
// ---------------------------------------------------------------------------
__global__ void bn_maxpool_pad(const _Float16* __restrict__ act,
                               _Float16* __restrict__ pooled,
                               const float* __restrict__ bnp, int Hc) {
  int Hp = Hc >> 1;
  int Wo = Hp + 2;
  int total = NIMG * NCH * Wo * Wo;
  int idx = blockIdx.x * blockDim.x + threadIdx.x;
  if (idx >= total) return;
  int oxp = idx % Wo; int t = idx / Wo;
  int oyp = t % Wo;   t /= Wo;
  int c   = t % NCH;  int img = t / NCH;
  _Float16 res = (_Float16)0.f;
  if (oxp >= 1 && oxp <= Hp && oyp >= 1 && oyp <= Hp) {
    int py = oyp - 1, px = oxp - 1;
    int g = (img < NSUP) ? 0 : 1;
    float scale = bnp[(g * 64 + c) * 2 + 0];
    float shift = bnp[(g * 64 + c) * 2 + 1];
    const _Float16* p = act + (size_t)(img * NCH + c) * Hc * Hc;
    float m = -3.4e38f;
    #pragma unroll
    for (int dy = 0; dy < 2; ++dy)
      #pragma unroll
      for (int dx = 0; dx < 2; ++dx) {
        float v = (float)p[(size_t)(2 * py + dy) * Hc + (2 * px + dx)] * scale + shift;
        m = fmaxf(m, v);
      }
    res = (_Float16)m;
  }
  pooled[idx] = res;
}

// ---------------------------------------------------------------------------
// 5b. BN apply + 2x2 maxpool -> fp32 features (layer 4, NCHW-flattened)
// ---------------------------------------------------------------------------
__global__ void bn_maxpool_feat(const _Float16* __restrict__ act,
                                float* __restrict__ feat,
                                const float* __restrict__ bnp) {
  const int Hc = 10, Hp = 5;
  int total = NIMG * NCH * Hp * Hp;
  int idx = blockIdx.x * blockDim.x + threadIdx.x;
  if (idx >= total) return;
  int px = idx % Hp; int t = idx / Hp;
  int py = t % Hp;   t /= Hp;
  int c  = t % NCH;  int img = t / NCH;
  int g = (img < NSUP) ? 0 : 1;
  float scale = bnp[(g * 64 + c) * 2 + 0];
  float shift = bnp[(g * 64 + c) * 2 + 1];
  const _Float16* p = act + (size_t)(img * NCH + c) * Hc * Hc;
  float m = -3.4e38f;
  #pragma unroll
  for (int dy = 0; dy < 2; ++dy)
    #pragma unroll
    for (int dx = 0; dx < 2; ++dx) {
      float v = (float)p[(size_t)(2 * py + dy) * Hc + (2 * px + dx)] * scale + shift;
      m = fmaxf(m, v);
    }
  feat[(size_t)img * DFEAT + c * (Hp * Hp) + py * Hp + px] = m;
}

// ---------------------------------------------------------------------------
// 6. Gram volumes via v_wmma_f32_16x16x4_f32 (A-frag == B-frag by symmetry)
// ---------------------------------------------------------------------------
__global__ __launch_bounds__(32) void gram_volA(const float* __restrict__ feat,
                                                float* __restrict__ volA) {
  int bw = blockIdx.x;            // 0..19
  int b = bw / 5, w = bw % 5;
  int lane = threadIdx.x;
  int row = lane & 15;
  bool valid = row < 4;
  const float* f0 = feat + (size_t)(b * 25 + w * 5) * DFEAT;
  const float* fr = feat + (size_t)(b * 25 + w * 5 + (valid ? row + 1 : 1)) * DFEAT;
  float msk = valid ? 1.f : 0.f;
  int ko = (lane < 16) ? 0 : 2;
  v8f acc = {};
  for (int k = 0; k < DFEAT; k += 4) {
    v2f a;
    a.x = (fr[k + ko]     - f0[k + ko])     * msk;
    a.y = (fr[k + ko + 1] - f0[k + ko + 1]) * msk;
    acc = __builtin_amdgcn_wmma_f32_16x16x4_f32(false, a, false, a, (short)0, acc, false, false);
  }
  __shared__ float G[16];
  if (lane < 4) {
    #pragma unroll
    for (int i = 0; i < 4; ++i) G[i * 4 + lane] = acc[i];
  }
  __syncthreads();
  if (lane == 0) {
    float g[16];
    #pragma unroll
    for (int i = 0; i < 16; ++i) g[i] = G[i];
    #pragma unroll
    for (int i = 0; i < 4; ++i) g[i * 4 + i] += 1.f;
    float det = 1.f;
    #pragma unroll
    for (int i = 0; i < 4; ++i) {
      float pv = g[i * 4 + i]; det *= pv; float inv = 1.f / pv;
      #pragma unroll
      for (int r = i + 1; r < 4; ++r) {
        float fk = g[r * 4 + i] * inv;
        #pragma unroll
        for (int c = 0; c < 4; ++c) g[r * 4 + c] -= fk * g[i * 4 + c];
      }
    }
    volA[bw] = fabsf(det);
  }
}

__global__ __launch_bounds__(32) void gram_logits(const float* __restrict__ feat,
                                                  const float* __restrict__ volA,
                                                  float* __restrict__ outLogits) {
  int x = blockIdx.x;             // 0..1499
  int b = x / 375; int r0 = x - b * 375;
  int q = r0 / 5;  int w = r0 - q * 5;
  int lane = threadIdx.x;
  int row = lane & 15;
  bool valid = row < 5;
  const float* fs = feat + (size_t)(b * 25 + w * 5 + (valid ? row : 0)) * DFEAT;
  const float* fq = feat + (size_t)(NSUP + b * 75 + q) * DFEAT;
  float msk = valid ? 1.f : 0.f;
  int ko = (lane < 16) ? 0 : 2;
  v8f acc = {};
  for (int k = 0; k < DFEAT; k += 4) {
    v2f a;
    a.x = (fs[k + ko]     - fq[k + ko])     * msk;
    a.y = (fs[k + ko + 1] - fq[k + ko + 1]) * msk;
    acc = __builtin_amdgcn_wmma_f32_16x16x4_f32(false, a, false, a, (short)0, acc, false, false);
  }
  __shared__ float G[25];
  if (lane < 5) {
    #pragma unroll
    for (int i = 0; i < 5; ++i) G[i * 5 + lane] = acc[i];
  }
  __syncthreads();
  if (lane == 0) {
    float g[25];
    #pragma unroll
    for (int i = 0; i < 25; ++i) g[i] = G[i];
    #pragma unroll
    for (int i = 0; i < 5; ++i) g[i * 5 + i] += 1.f;
    float det = 1.f;
    #pragma unroll
    for (int i = 0; i < 5; ++i) {
      float pv = g[i * 5 + i]; det *= pv; float inv = 1.f / pv;
      #pragma unroll
      for (int r = i + 1; r < 5; ++r) {
        float fk = g[r * 5 + i] * inv;
        #pragma unroll
        for (int c = 0; c < 5; ++c) g[r * 5 + c] -= fk * g[i * 5 + c];
      }
    }
    float d = fabsf(det) / volA[b * 5 + w];
    outLogits[(size_t)(b * 75 + q) * 5 + w] = -d;
  }
}

// ---------------------------------------------------------------------------
// 7. Row softmax over WAY=5
// ---------------------------------------------------------------------------
__global__ void softmax5(float* __restrict__ out) {
  int r = blockIdx.x * blockDim.x + threadIdx.x;
  if (r >= 300) return;
  float v[5];
  float m = -3.4e38f;
  #pragma unroll
  for (int j = 0; j < 5; ++j) { v[j] = out[r * 5 + j]; m = fmaxf(m, v[j]); }
  float s = 0.f;
  #pragma unroll
  for (int j = 0; j < 5; ++j) { v[j] = __expf(v[j] - m); s += v[j]; }
  float inv = 1.f / s;
  #pragma unroll
  for (int j = 0; j < 5; ++j) out[1500 + r * 5 + j] = v[j] * inv;
}

// ---------------------------------------------------------------------------
// Host-side orchestration
// ---------------------------------------------------------------------------
static inline size_t alignup(size_t x) { return (x + 255) & ~(size_t)255; }

extern "C" void kernel_launch(void* const* d_in, const int* in_sizes, int n_in,
                              void* d_out, int out_size, void* d_ws, size_t ws_size,
                              hipStream_t stream) {
  (void)in_sizes; (void)n_in; (void)out_size; (void)ws_size;
  const float* sup = (const float*)d_in[0];
  const float* smp = (const float*)d_in[1];
  const float* W[4]; const float* Bv[4]; const float* Gm[4]; const float* Bt[4];
  for (int i = 0; i < 4; ++i) {
    W[i]  = (const float*)d_in[2 + 4 * i];
    Bv[i] = (const float*)d_in[3 + 4 * i];
    Gm[i] = (const float*)d_in[4 + 4 * i];
    Bt[i] = (const float*)d_in[5 + 4 * i];
  }
  float* out = (float*)d_out;
  char* ws = (char*)d_ws;

  // workspace layout (padded activation buffers)
  size_t off = 0;
  _Float16* in0  = (_Float16*)(ws + off); off = alignup(off + (size_t)NIMG * 3 * 86 * 86 * 2);
  _Float16* wpk1 = (_Float16*)(ws + off); off = alignup(off + (size_t)4 * 1  * 32 * 16 * 2);
  _Float16* wpk2 = (_Float16*)(ws + off); off = alignup(off + (size_t)4 * 18 * 32 * 16 * 2);
  _Float16* wpk3 = (_Float16*)(ws + off); off = alignup(off + (size_t)4 * 18 * 32 * 16 * 2);
  _Float16* wpk4 = (_Float16*)(ws + off); off = alignup(off + (size_t)4 * 18 * 32 * 16 * 2);
  _Float16* actB = (_Float16*)(ws + off); off = alignup(off + (size_t)NIMG * NCH * 84 * 84 * 2);
  _Float16* p1   = (_Float16*)(ws + off); off = alignup(off + (size_t)NIMG * NCH * 44 * 44 * 2);
  _Float16* p2   = (_Float16*)(ws + off); off = alignup(off + (size_t)NIMG * NCH * 23 * 23 * 2);
  _Float16* p3   = (_Float16*)(ws + off); off = alignup(off + (size_t)NIMG * NCH * 12 * 12 * 2);
  float*    feat = (float*)   (ws + off); off = alignup(off + (size_t)NIMG * DFEAT * 4);
  float*    stats= (float*)   (ws + off); off = alignup(off + 256 * 4);
  float*    bnp  = (float*)   (ws + off); off = alignup(off + 256 * 4);
  float*    volA = (float*)   (ws + off); off = alignup(off + 32 * 4);

  // input conversion + weight packing
  {
    size_t total = (size_t)NIMG * 3 * 86 * 86;
    cvt_inputs<<<dim3((unsigned)((total + 255) / 256)), 256, 0, stream>>>(sup, smp, in0);
  }
  pack_w<<<dim3((4 * 1  * 32 * 16 + 255) / 256), 256, 0, stream>>>(W[0], wpk1, 3, 1);
  pack_w<<<dim3((4 * 18 * 32 * 16 + 255) / 256), 256, 0, stream>>>(W[1], wpk2, 64, 18);
  pack_w<<<dim3((4 * 18 * 32 * 16 + 255) / 256), 256, 0, stream>>>(W[2], wpk3, 64, 18);
  pack_w<<<dim3((4 * 18 * 32 * 16 + 255) / 256), 256, 0, stream>>>(W[3], wpk4, 64, 18);

  // ---- layer 1: 86x86 padded in -> 84x84 act -> 44x44 padded pool ----
  zero_f32<<<1, 256, 0, stream>>>(stats, 256);
  conv_wmma<84, 3, 1><<<dim3((84 * 84 + 63) / 64, NIMG), 128, 0, stream>>>(in0, wpk1, Bv[0], actB, stats);
  bn_finalize<<<1, 128, 0, stream>>>(stats, Gm[0], Bt[0], bnp, 84 * 84);
  bn_maxpool_pad<<<dim3((NIMG * NCH * 44 * 44 + 255) / 256), 256, 0, stream>>>(actB, p1, bnp, 84);

  // ---- layer 2: 44x44 padded in -> 42x42 act -> 23x23 padded pool ----
  zero_f32<<<1, 256, 0, stream>>>(stats, 256);
  conv_wmma<42, 64, 18><<<dim3((42 * 42 + 63) / 64, NIMG), 128, 0, stream>>>(p1, wpk2, Bv[1], actB, stats);
  bn_finalize<<<1, 128, 0, stream>>>(stats, Gm[1], Bt[1], bnp, 42 * 42);
  bn_maxpool_pad<<<dim3((NIMG * NCH * 23 * 23 + 255) / 256), 256, 0, stream>>>(actB, p2, bnp, 42);

  // ---- layer 3: 23x23 padded in -> 21x21 act -> 12x12 padded pool ----
  zero_f32<<<1, 256, 0, stream>>>(stats, 256);
  conv_wmma<21, 64, 18><<<dim3((21 * 21 + 63) / 64, NIMG), 128, 0, stream>>>(p2, wpk3, Bv[2], actB, stats);
  bn_finalize<<<1, 128, 0, stream>>>(stats, Gm[2], Bt[2], bnp, 21 * 21);
  bn_maxpool_pad<<<dim3((NIMG * NCH * 12 * 12 + 255) / 256), 256, 0, stream>>>(actB, p3, bnp, 21);

  // ---- layer 4: 12x12 padded in -> 10x10 act -> fp32 features ----
  zero_f32<<<1, 256, 0, stream>>>(stats, 256);
  conv_wmma<10, 64, 18><<<dim3((10 * 10 + 63) / 64, NIMG), 128, 0, stream>>>(p3, wpk4, Bv[3], actB, stats);
  bn_finalize<<<1, 128, 0, stream>>>(stats, Gm[3], Bt[3], bnp, 10 * 10);
  bn_maxpool_feat<<<dim3((NIMG * NCH * 5 * 5 + 255) / 256), 256, 0, stream>>>(actB, feat, bnp);

  // ---- distance head ----
  gram_volA<<<dim3(20), 32, 0, stream>>>(feat, volA);
  gram_logits<<<dim3(1500), 32, 0, stream>>>(feat, volA, out);
  softmax5<<<dim3(2), 256, 0, stream>>>(out);
}
